// STMambaEncoder_32796370273141
// MI455X (gfx1250) — matrix operand order repaired and checked by reading, not compiled, not run on hardware
//
#include <hip/hip_runtime.h>
#include <hip/hip_bf16.h>
#include <math.h>

// ---------------- model dims ----------------
#define NE   512
#define DI   1024
#define DSN  16      // d_state
#define DCV  4       // d_conv
#define DTR  32      // dt_rank
#define HW   196     // 14*14
#define NF   16      // B*T frames
#define LS   197     // spatial seq len (HW+1)
#define MS   (NF*LS) // 3152 spatial tokens
#define MT   16      // temporal tokens (2*8)

typedef __attribute__((ext_vector_type(16))) _Float16 v16h;
typedef __attribute__((ext_vector_type(8)))  _Float16 v8h;
typedef __attribute__((ext_vector_type(8)))  float    v8f;

// =====================================================================
// WMMA GEMM:  C[M,N] = act(A[M,K] @ W[N,K]^T + bias) (+= C)
//
// wave32. Block tile = 16 x (16*wpb), wpb = blockDim.x/32 (4 or 8).
// A tile (16x32 f16) staged in LDS once per k-step and reused by all
// waves; B tile (16*wpb x 32 f16) staged once; fragments fetched with
// 128-bit ds loads. One v_wmma_f32_16x16x32_f16 per wave per k-step.
// M % 16 == 0, N % (16*wpb) == 0, K % 32 == 0 (all guaranteed).
// act: 0=none, 1=gelu(exact erf), 2=silu, 3=softplus
// =====================================================================
__global__ __launch_bounds__(256) void k_wmma_gemm(
    const float* __restrict__ A, int lda,
    const float* __restrict__ W,
    const float* __restrict__ bias,
    float* __restrict__ C,
    int M, int N, int K, int act, int accum)
{
    __shared__ __align__(64) _Float16 As[16 * 32];        // 1 KB
    __shared__ __align__(64) _Float16 Bs[128 * 32];       // 8 KB (max wpb=8)

    const int nthreads = blockDim.x;
    const int wpb      = nthreads >> 5;                   // waves per block
    const int ntile    = wpb << 4;                        // block N extent
    const int tilesM   = M >> 4;
    const int tm       = blockIdx.x % tilesM;
    const int n0_block = (blockIdx.x / tilesM) * ntile;

    const int tid  = threadIdx.x;
    const int wave = tid >> 5;
    const int lane = tid & 31;
    const int r15  = lane & 15;
    const int hi   = lane >> 4;

    v8f acc = {};

    for (int k0 = 0; k0 < K; k0 += 32) {
        __syncthreads();                                  // previous compute done
        // ---- stage A tile: 16 rows x 32 cols, f32 -> f16 ----
        for (int i = tid; i < 16 * 32; i += nthreads)
            As[i] = (_Float16)A[(size_t)(tm * 16 + (i >> 5)) * lda + k0 + (i & 31)];
        // ---- stage B tile: ntile rows x 32 cols ----
        const int btot = ntile * 32;
        for (int i = tid; i < btot; i += nthreads)
            Bs[i] = (_Float16)W[(size_t)(n0_block + (i >> 5)) * K + k0 + (i & 31)];
        __syncthreads();                                  // tiles visible

        // ---- A fragment: lane = row, K 0..15 (lo half) / 16..31 (hi half), same order as B ----
        const int akoff = hi ? 16 : 0;
        v8h a_lo = *(const v8h*)&As[r15 * 32 + akoff];
        v8h a_hi = *(const v8h*)&As[r15 * 32 + akoff + 8];
        v16h a = __builtin_shufflevector(a_lo, a_hi,
                    0, 1, 2, 3, 4, 5, 6, 7, 8, 9, 10, 11, 12, 13, 14, 15);
        // ---- B fragment: lane = column, K 0..15 (lo half) / 16..31 (hi half) ----
        v16h b = *(const v16h*)&Bs[(wave * 16 + r15) * 32 + (hi ? 16 : 0)];

        acc = __builtin_amdgcn_wmma_f32_16x16x32_f16(
                  false, a, false, b, (short)0, acc, false, false);
    }

    const int   col = n0_block + wave * 16 + r15;
    const float bv  = bias ? bias[col] : 0.0f;
#pragma unroll
    for (int r = 0; r < 8; ++r) {
        const int row = tm * 16 + r + hi * 8;             // C/D layout: VGPR r -> M=r (+8 hi)
        float v = acc[r] + bv;
        if (act == 1)      v = 0.5f * v * (1.0f + erff(v * 0.70710678118654752f));
        else if (act == 2) v = v / (1.0f + __expf(-v));
        else if (act == 3) v = (v > 20.0f) ? v : __logf(1.0f + __expf(v));
        const size_t o = (size_t)row * N + col;
        if (accum) v += C[o];
        C[o] = v;
    }
}

// ---------------- stem: NCHW -> (pixel, channel) ----------------
__global__ __launch_bounds__(256) void k_stem_transpose(
    const float* __restrict__ x, float* __restrict__ xp)
{
    int idx = blockIdx.x * blockDim.x + threadIdx.x;      // over 16*256*196
    if (idx >= NF * 256 * HW) return;
    int p = idx % HW;
    int c = (idx / HW) & 255;
    int f = idx / (HW * 256);
    xp[((size_t)(f * HW + p)) * 256 + c] = x[idx];
}

// ---------------- assemble frame_token + stem tokens into h ----------------
__global__ __launch_bounds__(256) void k_assemble(
    const float* __restrict__ stem, const float* __restrict__ ft,
    float* __restrict__ h)
{
    int idx = blockIdx.x * blockDim.x + threadIdx.x;      // over 16*197*512
    if (idx >= NF * LS * NE) return;
    int e   = idx & (NE - 1);
    int tok = (idx >> 9) % LS;
    int f   = idx / (LS * NE);
    h[idx] = (tok == 0) ? ft[e]
                        : stem[((size_t)(f * HW + tok - 1)) * NE + e];
}

// ---------------- LayerNorm over NE=512, one wave per row ----------------
__global__ __launch_bounds__(256) void k_layernorm(
    const float* __restrict__ x, const float* __restrict__ w,
    const float* __restrict__ b, float* __restrict__ y, int rows)
{
    int wave = threadIdx.x >> 5;
    int lane = threadIdx.x & 31;
    int row  = blockIdx.x * 8 + wave;
    if (row >= rows) return;
    const float* xr = x + (size_t)row * NE;
    float s = 0.f, s2 = 0.f;
    for (int i = lane; i < NE; i += 32) { float v = xr[i]; s += v; s2 += v * v; }
    for (int m = 16; m > 0; m >>= 1) { s += __shfl_xor(s, m, 32); s2 += __shfl_xor(s2, m, 32); }
    float mu  = s * (1.0f / NE);
    float var = s2 * (1.0f / NE) - mu * mu;
    float rs  = rsqrtf(var + 1e-5f);
    float* yr = y + (size_t)row * NE;
    for (int i = lane; i < NE; i += 32)
        yr[i] = (xr[i] - mu) * rs * w[i] + b[i];
}

// ---------------- depthwise causal conv(4) + bias + SiLU ----------------
__global__ __launch_bounds__(256) void k_conv_silu(
    const float* __restrict__ xz, int xzld,
    const float* __restrict__ cw, const float* __restrict__ cb,
    float* __restrict__ out, int nseq, int Lseq)
{
    int idx = blockIdx.x * blockDim.x + threadIdx.x;
    int total = nseq * Lseq * DI;
    if (idx >= total) return;
    int d = idx & (DI - 1);
    int t = (idx >> 10) % Lseq;
    int s = idx / (DI * Lseq);
    float acc = cb[d];
#pragma unroll
    for (int j = 0; j < DCV; ++j) {
        int tt = t - (DCV - 1) + j;
        if (tt >= 0)
            acc += cw[d * DCV + j] * xz[(size_t)(s * Lseq + tt) * xzld + d];
    }
    out[(size_t)(s * Lseq + t) * DI + d] = acc / (1.0f + __expf(-acc));
}

// ---------------- selective scan + u*D + SiLU(z) gate ----------------
__global__ __launch_bounds__(256) void k_scan_gate(
    const float* __restrict__ xt,  const float* __restrict__ dt,
    const float* __restrict__ dbl, const float* __restrict__ A_log,
    const float* __restrict__ Dp,  const float* __restrict__ xz, int xzld,
    float* __restrict__ y, int nseq, int Lseq)
{
    int idx = blockIdx.x * blockDim.x + threadIdx.x;
    if (idx >= nseq * DI) return;
    int d = idx & (DI - 1);
    int s = idx >> 10;

    float Av[DSN];
#pragma unroll
    for (int n = 0; n < DSN; ++n) Av[n] = -__expf(A_log[d * DSN + n]);
    float hst[DSN];
#pragma unroll
    for (int n = 0; n < DSN; ++n) hst[n] = 0.f;
    const float Dd = Dp[d];

    for (int t = 0; t < Lseq; ++t) {
        const size_t row = (size_t)(s * Lseq + t);
        const float dtv = dt[row * DI + d];
        const float u   = xt[row * DI + d];
        const float du  = dtv * u;
        const float* Bm = dbl + row * (DTR + 2 * DSN) + DTR;
        const float* Cm = Bm + DSN;
        float acc = 0.f;
#pragma unroll
        for (int n = 0; n < DSN; ++n) {
            hst[n] = __expf(dtv * Av[n]) * hst[n] + du * Bm[n];
            acc += hst[n] * Cm[n];
        }
        acc += u * Dd;
        const float z = xz[row * xzld + DI + d];
        y[row * DI + d] = acc * (z / (1.0f + __expf(-z)));
    }
}

// ---------------- extract frame tokens into temporal buffer ----------------
__global__ __launch_bounds__(256) void k_extract_tokens(
    const float* __restrict__ h, float* __restrict__ h2)
{
    int idx = blockIdx.x * blockDim.x + threadIdx.x;   // over 16*512
    if (idx >= MT * NE) return;
    int e = idx & (NE - 1);
    int f = idx >> 9;
    h2[idx] = h[((size_t)f * LS) * NE + e];
}

// =====================================================================
// Host-side orchestration
// =====================================================================
static inline void launch_gemm(const float* A, int lda, const float* W,
                               const float* bias, float* C,
                               int M, int N, int K, int act, int accum,
                               hipStream_t s)
{
    int wpb   = (N >= 128) ? 8 : (N >> 4);   // waves per block (8, or 4 when N=64)
    int ntile = wpb << 4;
    int grid  = (M >> 4) * (N / ntile);
    k_wmma_gemm<<<grid, wpb * 32, 0, s>>>(A, lda, W, bias, C, M, N, K, act, accum);
}

static void run_block(void* const* din, int base, int layer,
                      float* h, int M, int nseq, int Lseq,
                      float* b_ln, float* b_xz, float* b_xt,
                      float* b_dbl, float* b_dt, float* b_y,
                      hipStream_t s)
{
    const float* ln1_w  = (const float*)din[base + 0]  + (size_t)layer * NE;
    const float* ln1_b  = (const float*)din[base + 1]  + (size_t)layer * NE;
    const float* ln2_w  = (const float*)din[base + 2]  + (size_t)layer * NE;
    const float* ln2_b  = (const float*)din[base + 3]  + (size_t)layer * NE;
    const float* in_w   = (const float*)din[base + 4]  + (size_t)layer * (2 * DI * NE);
    const float* conv_w = (const float*)din[base + 5]  + (size_t)layer * (DI * DCV);
    const float* conv_b = (const float*)din[base + 6]  + (size_t)layer * DI;
    const float* x_w    = (const float*)din[base + 7]  + (size_t)layer * ((DTR + 2 * DSN) * DI);
    const float* dt_w   = (const float*)din[base + 8]  + (size_t)layer * (DI * DTR);
    const float* dt_b   = (const float*)din[base + 9]  + (size_t)layer * DI;
    const float* A_log  = (const float*)din[base + 10] + (size_t)layer * (DI * DSN);
    const float* Dp     = (const float*)din[base + 11] + (size_t)layer * DI;
    const float* out_w  = (const float*)din[base + 12] + (size_t)layer * (NE * DI);
    const float* fc_w   = (const float*)din[base + 13] + (size_t)layer * (4 * NE * NE);
    const float* fc_b   = (const float*)din[base + 14] + (size_t)layer * (4 * NE);
    const float* proj_w = (const float*)din[base + 15] + (size_t)layer * (NE * 4 * NE);
    const float* proj_b = (const float*)din[base + 16] + (size_t)layer * NE;

    const int lnBlocks   = (M + 7) / 8;
    const int convTotal  = nseq * Lseq * DI;
    const int scanTotal  = nseq * DI;

    // ---- Mamba branch ----
    k_layernorm<<<lnBlocks, 256, 0, s>>>(h, ln1_w, ln1_b, b_ln, M);
    launch_gemm(b_ln, NE, in_w, nullptr, b_xz, M, 2 * DI, NE, 0, 0, s);   // xz = ln1 @ in_w^T
    k_conv_silu<<<(convTotal + 255) / 256, 256, 0, s>>>(b_xz, 2 * DI, conv_w, conv_b,
                                                        b_xt, nseq, Lseq);
    launch_gemm(b_xt, DI, x_w, nullptr, b_dbl, M, DTR + 2 * DSN, DI, 0, 0, s); // dbl = xt @ x_w^T
    launch_gemm(b_dbl, DTR + 2 * DSN, dt_w, dt_b, b_dt, M, DI, DTR, 3, 0, s); // dt = softplus(...)
    k_scan_gate<<<(scanTotal + 255) / 256, 256, 0, s>>>(b_xt, b_dt, b_dbl, A_log, Dp,
                                                        b_xz, 2 * DI, b_y, nseq, Lseq);
    launch_gemm(b_y, DI, out_w, nullptr, h, M, NE, DI, 0, 1, s);          // h += y @ out_w^T

    // ---- MLP branch (hidden reuses xz buffer) ----
    k_layernorm<<<lnBlocks, 256, 0, s>>>(h, ln2_w, ln2_b, b_ln, M);
    launch_gemm(b_ln, NE, fc_w, fc_b, b_xz, M, 4 * NE, NE, 1, 0, s);      // gelu
    launch_gemm(b_xz, 4 * NE, proj_w, proj_b, h, M, NE, 4 * NE, 0, 1, s); // h += proj
}

extern "C" void kernel_launch(void* const* d_in, const int* in_sizes, int n_in,
                              void* d_out, int out_size, void* d_ws, size_t ws_size,
                              hipStream_t stream)
{
    float* ws    = (float*)d_ws;
    float* b_ln  = ws;                                  // MS*NE
    float* b_xz  = b_ln  + (size_t)MS * NE;             // MS*2*DI
    float* b_xt  = b_xz  + (size_t)MS * 2 * DI;         // MS*DI
    float* b_dbl = b_xt  + (size_t)MS * DI;             // MS*64
    float* b_dt  = b_dbl + (size_t)MS * (DTR + 2 * DSN);// MS*DI
    float* b_y   = b_dt  + (size_t)MS * DI;             // MS*DI
    float* b_h   = b_y   + (size_t)MS * DI;             // MS*NE
    // stem temporaries alias the (not yet used) xz buffer
    float* b_xp   = b_xz;                               // 3136*256
    float* b_stem = b_xz + (size_t)(NF * HW) * 256;     // 3136*512

    const float* x  = (const float*)d_in[0];
    const float* cw = (const float*)d_in[1];
    const float* cb = (const float*)d_in[2];
    const float* ft = (const float*)d_in[3];

    // ---- stem: 1x1 conv as WMMA GEMM, then token assembly ----
    int tTotal = NF * 256 * HW;
    k_stem_transpose<<<(tTotal + 255) / 256, 256, 0, stream>>>(x, b_xp);
    launch_gemm(b_xp, 256, cw, cb, b_stem, NF * HW, NE, 256, 0, 0, stream);
    int aTotal = NF * LS * NE;
    k_assemble<<<(aTotal + 255) / 256, 256, 0, stream>>>(b_stem, ft, b_h);

    // ---- 4 spatial Mamba blocks (16 sequences of length 197) ----
    for (int i = 0; i < 4; ++i)
        run_block(d_in, 4, i, b_h, MS, NF, LS, b_ln, b_xz, b_xt, b_dbl, b_dt, b_y, stream);

    // ---- frame tokens -> temporal sequence; run in-place on d_out ----
    float* h2 = (float*)d_out;                          // (2,8,512) = 8192 f32
    k_extract_tokens<<<(MT * NE + 255) / 256, 256, 0, stream>>>(b_h, h2);

    // ---- 4 temporal Mamba blocks (2 sequences of length 8) ----
    for (int i = 0; i < 4; ++i)
        run_block(d_in, 21, i, h2, MT, 2, 8, b_ln, b_xz, b_xt, b_dbl, b_dt, b_y, stream);
}
